// cSE_13383118095125
// MI455X (gfx1250) — matrix operand order repaired and checked by reading, not compiled, not run on hardware
//
#include <hip/hip_runtime.h>
#include <hip/hip_bf16.h>
#include <stdint.h>

// ---------------------------------------------------------------------------
// Channel self-attention: energy = q q^T, attn = softmax(max - energy),
// out = attn q.  N=4, C=128, S=48^3=110592, all f32 in/out.
// f16 WMMA (f32 accumulate) keeps both GEMMs memory-bound on MI455X
// (AI = 43 FLOP/B vs ~13 FLOP/B machine balance for f32 WMMA).
// This revision: 3-deep software-pipelined fragment rotation so each WMMA
// waits on loads issued ~2 WMMAs earlier (partial dscnt waits) instead of
// a synchronous LDS round-trip per matmul.
// ---------------------------------------------------------------------------

typedef _Float16 v16h __attribute__((ext_vector_type(16)));
typedef _Float16 v8h  __attribute__((ext_vector_type(8)));
typedef float    v8f  __attribute__((ext_vector_type(8)));
typedef float    v4f  __attribute__((ext_vector_type(4)));
typedef __fp16   h2v  __attribute__((ext_vector_type(2)));

#define NBATCH 4
#define CCH    128
#define SDIM   110592            // 48^3
#define GRAM_CHUNKS 108          // k-chunks per batch -> 432 WGs
#define KSPAN  (SDIM / GRAM_CHUNKS)   // 1024 columns per WG
#define KC     64                // columns staged in LDS per iteration
#define ST     (KC + 8)          // padded LDS row stride (f16 elems)
#define SB     128               // s-columns per WG in the AV kernel
#define AST    (CCH + 8)         // padded stride for attention tile
#define QST    (CCH + 8)         // padded stride for transposed q tile

static __device__ inline unsigned pk2(float a, float b) {
    h2v h = __builtin_amdgcn_cvt_pkrtz(a, b);   // v_cvt_pk_rtz_f16_f32
    return __builtin_bit_cast(unsigned, h);
}

static __device__ inline v16h cat8(v8h lo, v8h hi) {
    return __builtin_shufflevector(lo, hi, 0,1,2,3,4,5,6,7,8,9,10,11,12,13,14,15);
}

// --------------------------- kernel 0: zero energy -------------------------
__global__ void zero_energy(float* __restrict__ e) {
    e[blockIdx.x * blockDim.x + threadIdx.x] = 0.0f;   // grid sized exactly
}

// --------------------------- kernel 1: Gram matrix -------------------------
// energy[n,i,j] += sum_k q[n,i,k] q[n,j,k] over this WG's k-span.
__global__ __launch_bounds__(256)
void gram_kernel(const float* __restrict__ x, float* __restrict__ energy) {
    __shared__ __align__(16) _Float16 tile[CCH * ST];   // 128 x 72 f16 = 18 KB

    const int n     = blockIdx.y;
    const int kbase = blockIdx.x * KSPAN;
    const int t     = threadIdx.x;
    const int w     = t >> 5;          // wave id (wave32!)
    const int lane  = t & 31;
    const int lrow  = lane & 15;
    const int h     = lane >> 4;

    const float* xb = x + (size_t)n * CCH * SDIM;

    v8f zero = {0.f,0.f,0.f,0.f,0.f,0.f,0.f,0.f};
    v8f acc[8];
    #pragma unroll
    for (int j = 0; j < 8; ++j) acc[j] = zero;

    v4f pref[8];
    #pragma unroll
    for (int i = 0; i < 8; ++i) {               // prefetch first chunk
        int e = t + i * 256, row = e >> 4, c4 = e & 15;
        pref[i] = *(const v4f*)(xb + (size_t)row * SDIM + kbase + c4 * 4);
    }

    for (int kc = kbase; kc < kbase + KSPAN; kc += KC) {
        // convert + store previously-fetched chunk into LDS
        #pragma unroll
        for (int i = 0; i < 8; ++i) {
            int e = t + i * 256, row = e >> 4, c4 = e & 15;
            uint2 p = make_uint2(pk2(pref[i][0], pref[i][1]),
                                 pk2(pref[i][2], pref[i][3]));
            *(uint2*)(&tile[row * ST + c4 * 4]) = p;
        }
        __syncthreads();

        // issue next chunk's global loads; they complete during the WMMAs
        if (kc + KC < kbase + KSPAN) {
            #pragma unroll
            for (int i = 0; i < 8; ++i) {
                int e = t + i * 256, row = e >> 4, c4 = e & 15;
                pref[i] = *(const v4f*)(xb + (size_t)row * SDIM + (kc + KC) + c4 * 4);
            }
        }

        #pragma unroll
        for (int k0 = 0; k0 < KC; k0 += 32) {
            const int arow = w * 16 + lrow;
            v16h a = cat8(*(const v8h*)(&tile[arow * ST + k0 + 16 * h]),
                          *(const v8h*)(&tile[arow * ST + k0 + 16 * h + 8]));

            // 3-deep rotating B-fragment pipeline: WMMA j consumes the slot
            // reloaded for j+3, so each WMMA waits only on loads issued two
            // WMMAs earlier (dscnt<=4, DS loads return in order).
            v16h bb[3];
            #pragma unroll
            for (int j = 0; j < 3; ++j) {
                const int brow = j * 16 + lrow;
                bb[j] = cat8(*(const v8h*)(&tile[brow * ST + k0 + 16 * h]),
                             *(const v8h*)(&tile[brow * ST + k0 + 16 * h + 8]));
            }
            #pragma unroll
            for (int j = 0; j < 8; ++j) {
                acc[j] = __builtin_amdgcn_wmma_f32_16x16x32_f16(
                    false, a, false, bb[j % 3], (short)0, acc[j], false, false);
                if (j + 3 < 8) {
                    const int brow = (j + 3) * 16 + lrow;
                    bb[j % 3] = cat8(
                        *(const v8h*)(&tile[brow * ST + k0 + 16 * h]),
                        *(const v8h*)(&tile[brow * ST + k0 + 16 * h + 8]));
                }
            }
        }
        __syncthreads();
    }

    float* en = energy + n * CCH * CCH;
    #pragma unroll
    for (int j = 0; j < 8; ++j) {
        #pragma unroll
        for (int v = 0; v < 8; ++v) {
            int row = w * 16 + 8 * h + v;          // C/D layout: M = 8h + v
            int col = j * 16 + lrow;               // N = lane&15
            atomicAdd(&en[row * CCH + col], acc[j][v]);
        }
    }
}

// --------------------------- kernel 2: softmax -----------------------------
// softmax(max(e) - e) == softmax(-e): attn = exp(min(e)-e) / sum.
__global__ __launch_bounds__(128) void softmax_kernel(const float* __restrict__ energy,
                                                      unsigned short* __restrict__ attn) {
    __shared__ float red[128];
    const int r = blockIdx.x;                      // n*128 + row
    const int t = threadIdx.x;

    float e = energy[r * CCH + t];
    red[t] = e;
    __syncthreads();
    for (int s = 64; s > 0; s >>= 1) {
        if (t < s) red[t] = fminf(red[t], red[t + s]);
        __syncthreads();
    }
    float mn = red[0];
    __syncthreads();

    float p = __expf(mn - e);
    red[t] = p;
    __syncthreads();
    for (int s = 64; s > 0; s >>= 1) {
        if (t < s) red[t] = red[t] + red[t + s];
        __syncthreads();
    }
    float a = p / red[0];

    union { _Float16 hh; unsigned short u; } cv;
    cv.hh = (_Float16)a;
    attn[r * CCH + t] = cv.u;
}

// --------------------------- kernel 3: out = attn * q ----------------------
__global__ __launch_bounds__(256)
void av_kernel(const float* __restrict__ x,
               const unsigned short* __restrict__ attn,
               float* __restrict__ out) {
    __shared__ __align__(16) _Float16 aT[CCH * AST];   // attention [i][m], 34 KB
    __shared__ __align__(16) _Float16 qT[SB * QST];    // q transposed [s][m], 34 KB

    const int n    = blockIdx.y;
    const int s0   = blockIdx.x * SB;
    const int t    = threadIdx.x;
    const int w    = t >> 5;
    const int lane = t & 31;
    const int lrow = lane & 15;
    const int h    = lane >> 4;

    // stage attention (already f16 in ws; 32 KB, L2-resident)
    const unsigned short* an = attn + n * CCH * CCH;
    #pragma unroll
    for (int i = 0; i < 8; ++i) {
        int e   = t + i * 256;                 // uint4 (8 x u16) index, 0..2047
        int row = e >> 4;
        int c8  = e & 15;
        uint4 v = *(const uint4*)(an + row * CCH + c8 * 8);
        *(uint4*)(&aT[row * AST + c8 * 8]) = v;
    }
    // stage q tile transposed via 4x4 register transpose:
    // coalesced float4 reads along s, packed uint2 stores along m.
    const float* xb = x + (size_t)n * CCH * SDIM;
    #pragma unroll
    for (int i = 0; i < 4; ++i) {
        int e  = t + i * 256;                  // 0..1023: 32x32 grid of 4x4 blocks
        int m4 = e >> 5;
        int s4 = e & 31;
        v4f vr[4];
        #pragma unroll
        for (int r = 0; r < 4; ++r)
            vr[r] = *(const v4f*)(xb + (size_t)(m4 * 4 + r) * SDIM + s0 + s4 * 4);
        #pragma unroll
        for (int c = 0; c < 4; ++c) {
            uint2 p = make_uint2(pk2(vr[0][c], vr[1][c]),
                                 pk2(vr[2][c], vr[3][c]));
            *(uint2*)(&qT[(s4 * 4 + c) * QST + m4 * 4]) = p;
        }
    }
    __syncthreads();

    v8f zero = {0.f,0.f,0.f,0.f,0.f,0.f,0.f,0.f};
    v8f acc[8];
    #pragma unroll
    for (int i = 0; i < 8; ++i) acc[i] = zero;

    #pragma unroll
    for (int k0 = 0; k0 < CCH; k0 += 32) {
        const int srow = w * 16 + lrow;
        v16h b = cat8(*(const v8h*)(&qT[srow * QST + k0 + 8 * h]),
                      *(const v8h*)(&qT[srow * QST + k0 + 16 + 8 * h]));

        // 3-deep rotating A-fragment pipeline (same scheme as gram_kernel)
        v16h aa[3];
        #pragma unroll
        for (int i = 0; i < 3; ++i) {
            const int arow = i * 16 + lrow;
            aa[i] = cat8(*(const v8h*)(&aT[arow * AST + k0 + 8 * h]),
                         *(const v8h*)(&aT[arow * AST + k0 + 16 + 8 * h]));
        }
        #pragma unroll
        for (int i = 0; i < 8; ++i) {
            acc[i] = __builtin_amdgcn_wmma_f32_16x16x32_f16(
                false, aa[i % 3], false, b, (short)0, acc[i], false, false);
            if (i + 3 < 8) {
                const int arow = (i + 3) * 16 + lrow;
                aa[i % 3] = cat8(
                    *(const v8h*)(&aT[arow * AST + k0 + 8 * h]),
                    *(const v8h*)(&aT[arow * AST + k0 + 16 + 8 * h]));
            }
        }
    }

    float* ob = out + (size_t)n * CCH * SDIM;
    #pragma unroll
    for (int i = 0; i < 8; ++i) {
        #pragma unroll
        for (int v = 0; v < 8; ++v) {
            int row = i * 16 + 8 * h + v;
            int col = s0 + w * 16 + lrow;
            ob[(size_t)row * SDIM + col] = acc[i][v];
        }
    }
}

// ---------------------------------------------------------------------------
extern "C" void kernel_launch(void* const* d_in, const int* in_sizes, int n_in,
                              void* d_out, int out_size, void* d_ws, size_t ws_size,
                              hipStream_t stream) {
    (void)in_sizes; (void)n_in; (void)out_size; (void)ws_size;

    const float* x = (const float*)d_in[0];
    float* out     = (float*)d_out;

    // ws layout: [0, 256KB) energy f32[4*128*128]; [256KB, 384KB) attn f16
    float* energy        = (float*)d_ws;
    unsigned short* attn = (unsigned short*)((char*)d_ws +
                              (size_t)NBATCH * CCH * CCH * sizeof(float));

    zero_energy<<<(NBATCH * CCH * CCH) / 256, 256, 0, stream>>>(energy);
    gram_kernel<<<dim3(GRAM_CHUNKS, NBATCH), 256, 0, stream>>>(x, energy);
    softmax_kernel<<<NBATCH * CCH, 128, 0, stream>>>(energy, attn);
    av_kernel<<<dim3(SDIM / SB, NBATCH), 256, 0, stream>>>(x, attn, out);
}
